// TCN1DAutoencoder_86706799772366
// MI455X (gfx1250) — 16-bit and 8-bit matrix operand order repaired and checked by reading, compiled, not run on hardware
//
#include <hip/hip_runtime.h>


#define Bn 16
#define Tn 4096
#define Cn 256
#define Ln 64
#define Kn 3

typedef __attribute__((ext_vector_type(16))) _Float16 v16h;
typedef __attribute__((ext_vector_type(8)))  float    v8f;
typedef __attribute__((ext_vector_type(8)))  int      v8i;

union AFrag16 { v16h v; uint4 q[2]; };
union Frag8   { v8i  v; uint4 q[2]; unsigned long long d[4]; };

__device__ __forceinline__ float reluf(float x) { return x > 0.f ? x : 0.f; }

__device__ __forceinline__ unsigned char f32_to_fp8(float x) {
#if defined(__has_builtin) && __has_builtin(__builtin_amdgcn_cvt_pk_fp8_f32)
  return (unsigned char)(__builtin_amdgcn_cvt_pk_fp8_f32(x, x, 0, false) & 0xff);
#else
  unsigned u = __float_as_uint(x);
  unsigned s = (u >> 24) & 0x80u;
  float ax = __uint_as_float(u & 0x7FFFFFFFu);
  if (ax >= 448.0f) return (unsigned char)(s | 0x7E);
  if (ax < 0.0009765625f) return (unsigned char)s;
  if (ax < 0.015625f) {
    int q = (int)(ax * 512.0f + 0.5f);
    if (q > 7) return (unsigned char)(s | 0x08);
    return (unsigned char)(s | (unsigned)q);
  }
  unsigned au = __float_as_uint(ax);
  au += 0x7FFFFu + ((au >> 20) & 1u);
  int e = (int)(au >> 23) - 127 + 7;
  unsigned m = (au >> 20) & 7u;
  if (e > 15 || (e == 15 && m == 7u)) return (unsigned char)(s | 0x7E);
  return (unsigned char)(s | ((unsigned)e << 3) | m);
#endif
}

__global__ void pack_koc(const float* __restrict__ src, _Float16* __restrict__ dst,
                         int total) {
  int i = blockIdx.x * 256 + threadIdx.x;
  if (i >= total) return;
  int li   = i / (Kn * Cn * Cn);
  int rem  = i % (Kn * Cn * Cn);
  int k    = rem / (Cn * Cn);
  int rem2 = rem % (Cn * Cn);
  int o = rem2 >> 8, c = rem2 & 255;
  dst[i] = (_Float16)src[(((size_t)(li * Cn + o) * Cn + c) * Kn) + k];
}

__global__ void pack_dec(const float* __restrict__ w1, const float* __restrict__ w2,
                         unsigned char* __restrict__ dst) {
  int i = blockIdx.x * 256 + threadIdx.x;
  int l = i >> 16, o = (i >> 8) & 255, c = i & 255;
  int blk = l >> 1;
  const float* s = (l & 1) ? w2 : w1;
  dst[i] = f32_to_fp8(s[(((size_t)(blk * Cn + o) * Cn + c) * Kn) + (Kn - 1)]);
}

__global__ void enc0a_kernel(const float* __restrict__ x, const float* __restrict__ w1,
                             const float* __restrict__ b1, _Float16* __restrict__ out) {
  size_t i = (size_t)blockIdx.x * 256 + threadIdx.x;
  int o = (int)(i & 255);
  int t = (int)((i >> 8) & (Tn - 1));
  int b = (int)(i >> 20);
  const float* xb = x + (size_t)b * Tn;
  float acc = b1[o];
#pragma unroll
  for (int k = 0; k < Kn; ++k) {
    int tt = t - (Kn - 1 - k);
    if (tt >= 0) acc += w1[o * Kn + k] * xb[tt];
  }
  out[(((size_t)(b * Tn + t)) << 8) + o] = (_Float16)reluf(acc);
}

template <bool PRED>
__device__ __forceinline__ void conv_mac(const _Float16* __restrict__ Hin,
                                         const _Float16* __restrict__ Wp,
                                         int b, int t0, int o, int hi, int m,
                                         int dil, v8f acc[4]) {
#pragma unroll
  for (int k = 0; k < Kn; ++k) {
    const int shift = (Kn - 1 - k) * dil;
    const _Float16* wrow = Wp + (((size_t)k << 16) + ((size_t)o << 8));
#pragma unroll
    for (int c = 0; c < 8; ++c) {
      const int c0 = c << 5;
      AFrag16 w;
      __builtin_prefetch(wrow + c0 + 32, 0, 1);
      w.q[0] = *(const uint4*)(wrow + c0 + 8 * hi);
      w.q[1] = *(const uint4*)(wrow + c0 + 16 + 8 * hi);
#pragma unroll
      for (int tt = 0; tt < 4; ++tt) {
        const int t = t0 + (tt << 4) + m - shift;
        AFrag16 a;
        if (PRED) {
          const bool vld = (t >= 0);
          const int tc = vld ? t : 0;
          const _Float16* arow = Hin + (((size_t)(b * Tn + tc)) << 8);
          a.q[0] = *(const uint4*)(arow + c0 + 8 * hi);
          a.q[1] = *(const uint4*)(arow + c0 + 16 + 8 * hi);
          if (!vld) { a.q[0] = make_uint4(0, 0, 0, 0); a.q[1] = make_uint4(0, 0, 0, 0); }
        } else {
          const _Float16* arow = Hin + (((size_t)(b * Tn + t)) << 8);
          a.q[0] = *(const uint4*)(arow + c0 + 8 * hi);
          a.q[1] = *(const uint4*)(arow + c0 + 16 + 8 * hi);
        }
        acc[tt] = __builtin_amdgcn_wmma_f32_16x16x32_f16(false, a.v, false, w.v,
                                                         (short)0, acc[tt], false, false);
      }
    }
  }
}

__global__ __launch_bounds__(32) void conv_gemm(
    const _Float16* __restrict__ Hin, const _Float16* __restrict__ Wp,
    const float* __restrict__ bias, const _Float16* skip,
    const float* __restrict__ x, const float* __restrict__ dw,
    const float* __restrict__ db, int skipmode, _Float16* Hout, int dil) {
  const int lane = threadIdx.x;
  const int t0 = blockIdx.x << 6;
  const int o0 = blockIdx.y << 4;
  const int b  = blockIdx.z;
  const int hi = lane >> 4;
  const int m  = lane & 15;
  const int o  = o0 + m;

  v8f acc[4] = {{}, {}, {}, {}};
  if (blockIdx.x == 0) {
    conv_mac<true>(Hin, Wp, b, t0, o, hi, m, dil, acc);
  } else {
    conv_mac<false>(Hin, Wp, b, t0, o, hi, m, dil, acc);
  }

  const float bval = bias[o];
#pragma unroll
  for (int tt = 0; tt < 4; ++tt) {
#pragma unroll
    for (int r = 0; r < 8; ++r) {
      const int tr = t0 + (tt << 4) + r + 8 * hi;
      float v = reluf(acc[tt][r] + bval);
      size_t oidx = (((size_t)(b * Tn + tr)) << 8) + o;
      if (skipmode == 1) {
        v = reluf(v + (float)skip[oidx]);
      } else if (skipmode == 2) {
        v = reluf(v + x[(size_t)b * Tn + tr] * dw[o] + db[o]);
      }
      Hout[oidx] = (_Float16)v;
    }
  }
}

__global__ void zero_kernel(float* __restrict__ p) {
  p[blockIdx.x * 256 + threadIdx.x] = 0.0f;
}

__global__ void mean_kernel(const _Float16* __restrict__ H, float* __restrict__ hm) {
  int b = blockIdx.x, chunk = blockIdx.y, c = threadIdx.x;
  int tbase = chunk * (Tn / 32);
  float s = 0.f;
  for (int t = tbase; t < tbase + Tn / 32; ++t)
    s += (float)H[(((size_t)(b * Tn + t)) << 8) + c];
  atomicAdd(&hm[b * Cn + c], s);
}

__global__ void latent_kernel(const float* __restrict__ hm, const float* __restrict__ W,
                              const float* __restrict__ bb, float* __restrict__ z) {
  int i = blockIdx.x * 256 + threadIdx.x;
  int b = i >> 6, l = i & 63;
  float s = bb[l];
  for (int c = 0; c < Cn; ++c) s += hm[b * Cn + c] * (1.0f / (float)Tn) * W[l * Cn + c];
  z[b * Ln + l] = s;
}

__global__ void decinit_kernel(const float* __restrict__ z, const float* __restrict__ W,
                               const float* __restrict__ bb, float* __restrict__ di) {
  int i = blockIdx.x * 256 + threadIdx.x;
  int b = i >> 8, c = i & 255;
  float s = bb[c];
  for (int l = 0; l < Ln; ++l) s += z[b * Ln + l] * W[c * Ln + l];
  di[i] = s;
}

__global__ __launch_bounds__(512) void decoder_kernel(
    const float* __restrict__ DIg, const unsigned char* __restrict__ Wd,
    const float* __restrict__ db1, const float* __restrict__ db2,
    const float* __restrict__ embW, const float* __restrict__ embB,
    const float* __restrict__ outW, const float* __restrict__ outB,
    float* __restrict__ recon) {
  __shared__ unsigned char bufA[2][Bn * Cn];
  __shared__ float HH[Bn * Cn];
  __shared__ float DI[Bn * Cn];
  __shared__ float DB[8 * Cn];
  __shared__ float EW[Cn], EB[Cn], OW[Cn];
  __shared__ float prevO[Bn], outpart[Bn];

  const int tid  = threadIdx.x;
  const int wv   = tid >> 5;
  const int lane = tid & 31;
  const int hi   = lane >> 4;
  const int col  = lane & 15;
  const int o    = wv * 16 + col;

  for (int i = tid; i < Bn * Cn; i += 512) DI[i] = DIg[i];
  for (int i = tid; i < Cn; i += 512) { EW[i] = embW[i]; EB[i] = embB[i]; OW[i] = outW[i]; }
  for (int i = tid; i < 4 * Cn; i += 512) {
    int blk = i >> 8, c = i & 255;
    DB[(2 * blk) * Cn + c]     = db1[i];
    DB[(2 * blk + 1) * Cn + c] = db2[i];
  }
  if (tid < Bn) prevO[tid] = 0.0f;

  Frag8 wfrag[8][4];
#pragma unroll
  for (int l = 0; l < 8; ++l) {
    const unsigned char* wrow = Wd + (((size_t)l) << 16) + (((size_t)o) << 8);
#pragma unroll
    for (int c = 0; c < 4; ++c) {
      wfrag[l][c].q[0] = *(const uint4*)(wrow + c * 64 + hi * 16);
      wfrag[l][c].q[1] = *(const uint4*)(wrow + c * 64 + 32 + hi * 16);
    }
  }
  const float outb = outB[0];
  __syncthreads();

  int cur = 0;
  for (int t = 0; t < Tn; ++t) {
#pragma unroll
    for (int r = 0; r < 8; ++r) {
      int i = tid + r * 512;
      int b = i >> 8, c = i & 255;
      float v = prevO[b] * EW[c] + EB[c] + DI[i];
      HH[i] = v;
      bufA[cur][i] = f32_to_fp8(v);
    }
    if (tid < Bn) outpart[tid] = 0.0f;
    __syncthreads();

#pragma unroll
    for (int blk = 0; blk < 4; ++blk) {
      {
        v8f acc = {};
#pragma unroll
        for (int c = 0; c < 4; ++c) {
          Frag8 a;
          const unsigned char* arow = &bufA[cur][(col << 8) + (c << 6) + hi * 16];
          a.d[0] = *(const unsigned long long*)(arow);
          a.d[1] = *(const unsigned long long*)(arow + 8);
          a.d[2] = *(const unsigned long long*)(arow + 32);
          a.d[3] = *(const unsigned long long*)(arow + 40);
          acc = __builtin_amdgcn_wmma_f32_16x16x64_fp8_fp8(a.v, wfrag[2 * blk][c].v,
                                                           (short)0, acc, false, false);
        }
        const float bv = DB[(2 * blk) * Cn + o];
        const int nxt = cur ^ 1;
#pragma unroll
        for (int r = 0; r < 8; ++r) {
          int mrow = r + 8 * hi;
          bufA[nxt][(mrow << 8) + o] = f32_to_fp8(reluf(acc[r] + bv));
        }
        __syncthreads();
        cur ^= 1;
      }
      {
        v8f acc = {};
#pragma unroll
        for (int c = 0; c < 4; ++c) {
          Frag8 a;
          const unsigned char* arow = &bufA[cur][(col << 8) + (c << 6) + hi * 16];
          a.d[0] = *(const unsigned long long*)(arow);
          a.d[1] = *(const unsigned long long*)(arow + 8);
          a.d[2] = *(const unsigned long long*)(arow + 32);
          a.d[3] = *(const unsigned long long*)(arow + 40);
          acc = __builtin_amdgcn_wmma_f32_16x16x64_fp8_fp8(a.v, wfrag[2 * blk + 1][c].v,
                                                           (short)0, acc, false, false);
        }
        const float bv = DB[(2 * blk + 1) * Cn + o];
        const int nxt = cur ^ 1;
#pragma unroll
        for (int r = 0; r < 8; ++r) {
          int mrow = r + 8 * hi;
          float h2 = reluf(acc[r] + bv);
          float hn = reluf(h2 + HH[(mrow << 8) + o]);
          HH[(mrow << 8) + o] = hn;
          bufA[nxt][(mrow << 8) + o] = f32_to_fp8(hn);
        }
        __syncthreads();
        cur ^= 1;
      }
    }

#pragma unroll
    for (int r = 0; r < 8; ++r) {
      int mrow = r + 8 * hi;
      float p = HH[(mrow << 8) + o] * OW[o];
#pragma unroll
      for (int d = 1; d < 16; d <<= 1) p += __shfl_xor(p, d, 32);
      if (col == 0) atomicAdd(&outpart[mrow], p);
    }
    __syncthreads();
    if (tid < Bn) {
      float oo = outpart[tid] + outb;
      prevO[tid] = oo;
      recon[(size_t)tid * Tn + t] = oo;
    }
    __syncthreads();
  }
}

extern "C" void kernel_launch(void* const* d_in, const int* in_sizes, int n_in,
                              void* d_out, int out_size, void* d_ws, size_t ws_size,
                              hipStream_t stream) {
  (void)in_sizes; (void)n_in; (void)out_size; (void)ws_size;
  const float* x           = (const float*)d_in[0];
  const float* enc0_w1     = (const float*)d_in[1];
  const float* enc0_b1     = (const float*)d_in[2];
  const float* enc0_w2     = (const float*)d_in[3];
  const float* enc0_b2     = (const float*)d_in[4];
  const float* enc0_down_w = (const float*)d_in[5];
  const float* enc0_down_b = (const float*)d_in[6];
  const float* enc_w1      = (const float*)d_in[7];
  const float* enc_b1      = (const float*)d_in[8];
  const float* enc_w2      = (const float*)d_in[9];
  const float* enc_b2      = (const float*)d_in[10];
  const float* to_latent_w = (const float*)d_in[11];
  const float* to_latent_b = (const float*)d_in[12];
  const float* lat2dec_w   = (const float*)d_in[13];
  const float* lat2dec_b   = (const float*)d_in[14];
  const float* embed_w     = (const float*)d_in[15];
  const float* embed_b     = (const float*)d_in[16];
  const float* dec_w1      = (const float*)d_in[17];
  const float* dec_b1      = (const float*)d_in[18];
  const float* dec_w2      = (const float*)d_in[19];
  const float* dec_b2      = (const float*)d_in[20];
  const float* out_w       = (const float*)d_in[21];
  const float* out_b       = (const float*)d_in[22];

  char* ws = (char*)d_ws;
  size_t off = 0;
  auto take = [&](size_t bytes) -> char* {
    char* p = ws + off;
    off = (off + bytes + 255) & ~(size_t)255;
    return p;
  };
  _Float16* bufX = (_Float16*)take((size_t)Bn * Tn * Cn * 2);
  _Float16* bufY = (_Float16*)take((size_t)Bn * Tn * Cn * 2);
  _Float16* Wp0  = (_Float16*)take((size_t)Kn * Cn * Cn * 2);
  _Float16* Wp1  = (_Float16*)take((size_t)3 * Kn * Cn * Cn * 2);
  _Float16* Wp2  = (_Float16*)take((size_t)3 * Kn * Cn * Cn * 2);
  unsigned char* Wd = (unsigned char*)take((size_t)8 * Cn * Cn);
  float* hm  = (float*)take((size_t)Bn * Cn * 4);
  float* DIg = (float*)take((size_t)Bn * Cn * 4);

  float* recon = (float*)d_out;
  float* zout  = (float*)d_out + (size_t)Bn * Tn;

  pack_koc<<<(Kn * Cn * Cn + 255) / 256, 256, 0, stream>>>(enc0_w2, Wp0, Kn * Cn * Cn);
  pack_koc<<<(3 * Kn * Cn * Cn + 255) / 256, 256, 0, stream>>>(enc_w1, Wp1, 3 * Kn * Cn * Cn);
  pack_koc<<<(3 * Kn * Cn * Cn + 255) / 256, 256, 0, stream>>>(enc_w2, Wp2, 3 * Kn * Cn * Cn);
  pack_dec<<<(8 * Cn * Cn) / 256, 256, 0, stream>>>(dec_w1, dec_w2, Wd);

  enc0a_kernel<<<(Bn * Tn * Cn) / 256, 256, 0, stream>>>(x, enc0_w1, enc0_b1, bufY);
  dim3 cg(Tn / 64, Cn / 16, Bn);
  conv_gemm<<<cg, 32, 0, stream>>>(bufY, Wp0, enc0_b2, nullptr, x, enc0_down_w,
                                   enc0_down_b, 2, bufX, 1);
  for (int i = 0; i < 3; ++i) {
    int dil = 2 << i;
    conv_gemm<<<cg, 32, 0, stream>>>(bufX, Wp1 + (size_t)i * Kn * Cn * Cn,
                                     enc_b1 + i * Cn, nullptr, nullptr, nullptr,
                                     nullptr, 0, bufY, dil);
    conv_gemm<<<cg, 32, 0, stream>>>(bufY, Wp2 + (size_t)i * Kn * Cn * Cn,
                                     enc_b2 + i * Cn, bufX, nullptr, nullptr,
                                     nullptr, 1, bufX, dil);
  }

  zero_kernel<<<Bn, 256, 0, stream>>>(hm);
  mean_kernel<<<dim3(Bn, 32), Cn, 0, stream>>>(bufX, hm);
  latent_kernel<<<(Bn * Ln) / 256, 256, 0, stream>>>(hm, to_latent_w, to_latent_b, zout);
  decinit_kernel<<<(Bn * Cn) / 256, 256, 0, stream>>>(zout, lat2dec_w, lat2dec_b, DIg);

  decoder_kernel<<<1, 512, 0, stream>>>(DIg, Wd, dec_b1, dec_b2, embed_w, embed_b,
                                        out_w, out_b, recon);
}
